// SpikingAttentionSVC_9234179687031
// MI455X (gfx1250) — matrix operand order repaired and checked by reading, not compiled, not run on hardware
//
#include <hip/hip_runtime.h>
#include <hip/hip_bf16.h>

#define VOCAB     50257
#define SEQ_LEN   8192
#define K_WINNERS 5

typedef __attribute__((ext_vector_type(16))) _Float16 v16h;
typedef __attribute__((ext_vector_type(8)))  float    v8f;

// ---------------------------------------------------------------------------
// Kernel 1: zero the histogram bins (workspace) and the 3 svc accumulators.
// ---------------------------------------------------------------------------
__global__ void init_kernel(int* __restrict__ counts, float* __restrict__ svc_out) {
    int i = blockIdx.x * blockDim.x + threadIdx.x;
    if (i < VOCAB) counts[i] = 0;
    if (i < 3)     svc_out[i] = 0.0f;
}

// ---------------------------------------------------------------------------
// Kernel 2: histogram of tokens. (The LIF neuron spikes on every visit since
// v = 0.7*mem + 1.0 >= 1.0 always, so spike_counts == occurrence counts.)
// ---------------------------------------------------------------------------
__global__ void hist_kernel(const int* __restrict__ toks, int* __restrict__ counts) {
    int i = blockIdx.x * blockDim.x + threadIdx.x;
    if (i < SEQ_LEN) {
        int t = toks[i];
        if (t >= 0 && t < VOCAB) atomicAdd(&counts[t], 1);
    }
}

// ---------------------------------------------------------------------------
// Kernel 3: top-5 with jax.lax.top_k tie-breaking (higher count first; ties ->
// lower index). Key = (count << 32) | (0xFFFFFFFF - index); max key wins.
// Single block, 5 exclusion passes over the 50257 bins (L2-resident, ~us).
// ---------------------------------------------------------------------------
__global__ void topk_kernel(const int* __restrict__ counts,
                            int* __restrict__ top_idx,
                            float* __restrict__ top_val) {
    __shared__ unsigned long long red[256];
    __shared__ int chosen[K_WINNERS];
    const int tid = threadIdx.x;
    for (int r = 0; r < K_WINNERS; ++r) {
        unsigned long long best = 0ULL;
        for (int i = tid; i < VOCAB; i += 256) {
            bool skip = false;
            for (int q = 0; q < r; ++q) skip |= (chosen[q] == i);
            if (skip) continue;
            unsigned long long key =
                ((unsigned long long)(unsigned)counts[i] << 32) |
                (unsigned long long)(0xFFFFFFFFu - (unsigned)i);
            if (key > best) best = key;
        }
        red[tid] = best;
        __syncthreads();
        for (int s = 128; s > 0; s >>= 1) {
            if (tid < s) { if (red[tid + s] > red[tid]) red[tid] = red[tid + s]; }
            __syncthreads();
        }
        if (tid == 0) {
            unsigned long long k = red[0];
            int idx = (int)(0xFFFFFFFFu - (unsigned)(k & 0xFFFFFFFFull));
            chosen[r]  = idx;
            top_idx[r] = idx;
            top_val[r] = (float)(unsigned)(k >> 32);
        }
        __syncthreads();
    }
}

// ---------------------------------------------------------------------------
// Kernel 4: gains. Default 1.0, active -> 0.6, top-5 with val>0 -> 1.5.
// ---------------------------------------------------------------------------
__global__ void gains_kernel(const int* __restrict__ counts,
                             const int* __restrict__ top_idx,
                             const float* __restrict__ top_val,
                             float* __restrict__ gains) {
    int i = blockIdx.x * blockDim.x + threadIdx.x;
    if (i >= VOCAB) return;
    int c = counts[i];
    float g = (c > 0) ? 0.6f : 1.0f;
#pragma unroll
    for (int r = 0; r < K_WINNERS; ++r)
        if (top_idx[r] == i && top_val[r] > 0.0f) g = 1.5f;
    gains[i] = g;
}

// ---------------------------------------------------------------------------
// Kernel 5: svc_spikes[c] = sum_s counts[tok[s]] * (mask[s,c] > 0) via
// V_WMMA_F32_16X16X32_F16, fully branchless fragment construction.
// Counts are small integers (exact in f16; exact f32 accumulation, totals
// << 2^24) so any summation order is bit-exact vs the f32 reference.
//
// Fragment layouts (ISA 7.12.2, wave32):
//   A 16x32 f16, broadcast along M: A[m,k] = counts[tok[base+k]]
//     lane<16: K in {0..7, 16..23}; lane>=16: K in {8..15, 24..31}
//   B 32x16 f16: B[k,n] = (n<3) ? (mask[base+k, n] > 0) : 0
//     lanes 0..15 hold K={0..7,16..23} (n = lane&15); lanes 16..31 hold K={8..15,24..31}
//   D f32 16x16: D[0,n] sits in lanes 0..2, VGPR0 -> 3 atomicAdds per wave.
// Two tiles (2 WMMAs, independent accumulators) per loop iteration for ILP.
// ---------------------------------------------------------------------------
__device__ __forceinline__ void build_tile(const int* __restrict__ toks,
                                           const float* __restrict__ svc_mask,
                                           const int* __restrict__ counts,
                                           int base, int hiA, int hiB,
                                           int cidx, bool ncol,
                                           v16h& a, v16h& b) {
    // Tokens for this lane's A fragment: two aligned groups of 8 -> 4x int4.
    const int4* tp = (const int4*)(toks + base + hiA);        // K = hiA..hiA+7
    int4 t0 = tp[0];
    int4 t1 = tp[1];
    const int4* tq = (const int4*)(toks + base + 16 + hiA);   // K = 16+hiA..23+hiA
    int4 t2 = tq[0];
    int4 t3 = tq[1];
    int tk[16] = { t0.x, t0.y, t0.z, t0.w, t1.x, t1.y, t1.z, t1.w,
                   t2.x, t2.y, t2.z, t2.w, t3.x, t3.y, t3.z, t3.w };
#pragma unroll
    for (int j = 0; j < 16; ++j) {
        // Branchless clamp (tokens are in-range; matches JAX gather clamping).
        int tok = min(max(tk[j], 0), VOCAB - 1);
        a[j] = (_Float16)(float)counts[tok];
        // Mask column: every lane loads a valid address, selects {0,1} after.
        int sB = base + hiA + j + ((j >= 8) ? 8 : 0);
        float mv = svc_mask[sB * 3 + cidx];
        b[j] = (ncol && mv > 0.0f) ? (_Float16)1.0f : (_Float16)0.0f;
    }
}

__global__ void svc_wmma_kernel(const int* __restrict__ toks,
                                const float* __restrict__ svc_mask,
                                const int* __restrict__ counts,
                                float* __restrict__ svc_out) {
    const int lane = threadIdx.x & 31;
    const int wave = threadIdx.x >> 5;                  // 0..7 (256 threads)
    const int tiles_per_wave = (SEQ_LEN / 32) / 8;      // 32
    const int n    = lane & 15;
    const int cidx = (n < 2) ? n : 2;                   // safe mask column
    const bool ncol = (n < 3);
    const int hiA = (lane >= 16) ? 8 : 0;
    const int hiB = (lane >= 16) ? 16 : 0;

    v8f acc0 = {};
    v8f acc1 = {};
    const int first = wave * tiles_per_wave;
    for (int t = 0; t < tiles_per_wave; t += 2) {
        v16h a0, b0, a1, b1;
        build_tile(toks, svc_mask, counts, (first + t) * 32,     hiA, hiB, cidx, ncol, a0, b0);
        build_tile(toks, svc_mask, counts, (first + t + 1) * 32, hiA, hiB, cidx, ncol, a1, b1);
        acc0 = __builtin_amdgcn_wmma_f32_16x16x32_f16(
            false, a0, false, b0, (short)0, acc0, false, false);
        acc1 = __builtin_amdgcn_wmma_f32_16x16x32_f16(
            false, a1, false, b1, (short)0, acc1, false, false);
    }
    // D[0, n] for n=0..2 lives in lanes 0..2, VGPR 0 of each accumulator.
    float part = acc0[0] + acc1[0];
    if (lane < 3) atomicAdd(&svc_out[lane], part);
}

// ---------------------------------------------------------------------------
// Launch: d_in[0]=tokens (int[8192]), d_in[1]=svc_mask (f32[8192,3]).
// d_out = [gains f32[50257] | svc f32[3]]. d_ws: counts + top-5 scratch.
// ---------------------------------------------------------------------------
extern "C" void kernel_launch(void* const* d_in, const int* in_sizes, int n_in,
                              void* d_out, int out_size, void* d_ws, size_t ws_size,
                              hipStream_t stream) {
    const int*   toks     = (const int*)d_in[0];
    const float* svc_mask = (const float*)d_in[1];
    float* out   = (float*)d_out;
    float* svc   = out + VOCAB;

    int* counts = (int*)d_ws;
    size_t off = (((size_t)VOCAB * sizeof(int)) + 255) & ~(size_t)255;
    int*   top_idx = (int*)((char*)d_ws + off);
    float* top_val = (float*)((char*)d_ws + off + 8 * sizeof(int));

    const int B = 256;
    init_kernel<<<(VOCAB + B - 1) / B, B, 0, stream>>>(counts, svc);
    hist_kernel<<<(SEQ_LEN + B - 1) / B, B, 0, stream>>>(toks, counts);
    topk_kernel<<<1, B, 0, stream>>>(counts, top_idx, top_val);
    gains_kernel<<<(VOCAB + B - 1) / B, B, 0, stream>>>(counts, top_idx, top_val, out);
    svc_wmma_kernel<<<1, B, 0, stream>>>(toks, svc_mask, counts, svc);
}
